// DA_RNN_MultiTask_58377195487824
// MI455X (gfx1250) — compile-verified, matrix operand order checked by reading, not run on hardware
//
#include <hip/hip_runtime.h>

#define BQ 2048
#define TT 128
#define FF 32
#define HE 128
#define HD 128
#define G4 512   // 4*HE gates
#define NT_THREADS 256   // 8 waves

typedef __attribute__((ext_vector_type(16))) __bf16 v16bf;
typedef __attribute__((ext_vector_type(8)))  float  v8f;

// ---------------------------------------------------------------------------
// WMMA helpers
// ---------------------------------------------------------------------------
// Load a 16x32 bf16 A-fragment (or 32x16 B-fragment from an [N][K] transposed
// weight array) from row-major storage with row stride `stride` (elements).
// CDNA5 16-bit A layout: lane l holds row (l&15), K-chunks
// [(l>>4)*8, +8) and [(l>>4)*8+16, +8)  -> two 16-byte loads.
__device__ __forceinline__ v16bf ld_frag(const __bf16* p0, int stride)
{
    const int l = threadIdx.x & 31;
    const __bf16* p = p0 + (l & 15) * stride + ((l >> 4) << 3);
    union { uint4 u[2]; v16bf v; } un;
    un.u[0] = *(const uint4*)(p);
    un.u[1] = *(const uint4*)(p + 16);
    return un.v;
}

__device__ __forceinline__ v8f wmma_bf16(v16bf a, v16bf b, v8f c)
{
    return __builtin_amdgcn_wmma_f32_16x16x32_bf16(
        /*neg_a=*/false, a, /*neg_b=*/false, b,
        /*c_mod=*/(short)0, c, /*reuse_a=*/false, /*reuse_b=*/false);
}

// Fast activations: v_exp_f32 + v_rcp_f32, no IEEE division sequences.
__device__ __forceinline__ float frcp(float x) { return __builtin_amdgcn_rcpf(x); }
__device__ __forceinline__ float fsig(float x) { return frcp(1.0f + __expf(-x)); }
__device__ __forceinline__ float ftanh(float x)
{
    float e = __expf(2.0f * x);
    return 1.0f - 2.0f * frcp(e + 1.0f);
}

__device__ __forceinline__ void cp_bf16(__bf16* dst, const __bf16* src, int nelem)
{
    uint4* d = (uint4*)dst; const uint4* s = (const uint4*)src;
    for (int i = threadIdx.x; i < (nelem >> 3); i += blockDim.x) d[i] = s[i];
}
__device__ __forceinline__ void cp_f32(float* dst, const float* src, int nelem)
{
    for (int i = threadIdx.x; i < nelem; i += blockDim.x) dst[i] = src[i];
}

// ---------------------------------------------------------------------------
// Prep: W (K x N, f32 row-major) -> WT ([N][K], bf16)   (weights -> L2, tiny)
// ---------------------------------------------------------------------------
__global__ void transpose_bf16_kernel(const float* __restrict__ W,
                                      __bf16* __restrict__ WT, int K, int N)
{
    int i = blockIdx.x * blockDim.x + threadIdx.x;
    if (i < K * N) {
        int k = i / N, n = i % N;
        WT[n * K + k] = (__bf16)W[i];
    }
}

// ---------------------------------------------------------------------------
// Encoder: persistent kernel, 16 batch rows per WG, loops t = 0..T-1.
// 8 waves: attention GEMM1 -> 1 N-tile/wave; gate GEMM -> 64 cols/wave.
// ---------------------------------------------------------------------------
__global__ __launch_bounds__(NT_THREADS, 1)
void enc_kernel(const float* __restrict__ X,
                const __bf16* __restrict__ gWihT,   // [512][32]
                const __bf16* __restrict__ gWhhT,   // [512][128]
                const __bf16* __restrict__ gW1T,    // [128][256]
                const __bf16* __restrict__ gW2T,    // [32][128]
                const float*  __restrict__ g_b,     // [512]
                const float*  __restrict__ g_ab1,   // [128]
                const float*  __restrict__ g_ab2,   // [32]
                __bf16* __restrict__ enc_hs)        // [B][T][128] bf16
{
    __shared__ __align__(16) __bf16 sWhhT[G4 * HE];       // 131072 B
    __shared__ __align__(16) __bf16 sWihT[G4 * FF];       //  32768 B
    __shared__ __align__(16) __bf16 sW1T [HE * 2 * HE];   //  65536 B
    __shared__ __align__(16) __bf16 sW2T [FF * HE];       //   8192 B
    __shared__ __align__(16) __bf16 hcat [16 * 2 * HE];   // [h|c] bf16
    __shared__ __align__(16) __bf16 tnh  [16 * HE];
    __shared__ __align__(16) __bf16 ax   [16 * FF];
    __shared__ __align__(16) float  cst  [16 * HE];       // cell state f32
    __shared__ __align__(16) float  gat  [16 * G4];
    __shared__ __align__(16) float  sc   [16 * FF];
    __shared__ __align__(16) float  sb   [G4];
    __shared__ __align__(16) float  sab1 [HE];
    __shared__ __align__(16) float  sab2 [FF];

    const int tid = threadIdx.x;
    const int w   = tid >> 5;          // 0..7
    const int l   = tid & 31;
    const int b0  = blockIdx.x * 16;

    cp_bf16(sWhhT, gWhhT, G4 * HE);
    cp_bf16(sWihT, gWihT, G4 * FF);
    cp_bf16(sW1T,  gW1T,  HE * 2 * HE);
    cp_bf16(sW2T,  gW2T,  FF * HE);
    cp_f32(sb,   g_b,   G4);
    cp_f32(sab1, g_ab1, HE);
    cp_f32(sab2, g_ab2, FF);
    for (int i = tid; i < 16 * 2 * HE; i += NT_THREADS) hcat[i] = (__bf16)0.0f;
    for (int i = tid; i < 16 * HE;     i += NT_THREADS) cst[i]  = 0.0f;
    __syncthreads();

    const int mrow = 8 * (l >> 4);
    const int ncol = l & 15;
    const v8f vzero = {0.f,0.f,0.f,0.f,0.f,0.f,0.f,0.f};

    for (int t = 0; t < TT; ++t) {
        // prefetch next step's x_t tile while we compute (global_prefetch_b8)
        if (t + 1 < TT && tid < 16)
            __builtin_prefetch(&X[((size_t)(b0 + tid) * TT + (t + 1)) * FF], 0, 0);

        // ---- Phase A: pre = tanh([h,c] @ W1 + b1), 16x256 @ 256x128 ----
        {
            const int nt = w;   // one 16-col tile per wave
            v8f acc = vzero;
#pragma unroll
            for (int kc = 0; kc < 8; ++kc) {
                v16bf a = ld_frag(hcat + kc * 32, 2 * HE);
                v16bf b = ld_frag(sW1T + nt * 16 * (2 * HE) + kc * 32, 2 * HE);
                acc = wmma_bf16(a, b, acc);
            }
            int n = nt * 16 + ncol;
#pragma unroll
            for (int r = 0; r < 8; ++r)
                tnh[(mrow + r) * HE + n] = (__bf16)ftanh(acc[r] + sab1[n]);
        }
        __syncthreads();

        // ---- Phase B: scores = tnh @ W2 + b2, 16x128 @ 128x32 ----
        if (w < 2) {
            const int nt = w;
            v8f acc = vzero;
#pragma unroll
            for (int kc = 0; kc < 4; ++kc) {
                v16bf a = ld_frag(tnh + kc * 32, HE);
                v16bf b = ld_frag(sW2T + nt * 16 * HE + kc * 32, HE);
                acc = wmma_bf16(a, b, acc);
            }
            int n = nt * 16 + ncol;
#pragma unroll
            for (int r = 0; r < 8; ++r)
                sc[(mrow + r) * FF + n] = acc[r] + sab2[n];
        }
        __syncthreads();

        // ---- Phase C: softmax over F per row, then ax = alpha * x_t ----
        if (tid < 16) {
            float mx = -1e30f;
            for (int f = 0; f < FF; ++f) mx = fmaxf(mx, sc[tid * FF + f]);
            float sum = 0.0f;
            for (int f = 0; f < FF; ++f) {
                float e = __expf(sc[tid * FF + f] - mx);
                sc[tid * FF + f] = e; sum += e;
            }
            float inv = frcp(sum);
            for (int f = 0; f < FF; ++f) sc[tid * FF + f] *= inv;
        }
        __syncthreads();
        for (int i = tid; i < 16 * FF; i += NT_THREADS) {
            int m = i >> 5, f = i & 31;
            float xv = X[((size_t)(b0 + m) * TT + t) * FF + f];
            ax[i] = (__bf16)(sc[i] * xv);
        }
        __syncthreads();

        // ---- Phase D: gates = ax @ Wih + h @ Whh  (each wave owns 64 N) ----
        {
            const int nbase = w * 64;
            v8f acc[4];
#pragma unroll
            for (int i = 0; i < 4; ++i) acc[i] = vzero;
            v16bf axf = ld_frag(ax, FF);
#pragma unroll
            for (int nt = 0; nt < 4; ++nt) {
                v16bf b = ld_frag(sWihT + (nbase + nt * 16) * FF, FF);
                acc[nt] = wmma_bf16(axf, b, acc[nt]);
            }
            v16bf ah[4];
#pragma unroll
            for (int kc = 0; kc < 4; ++kc) ah[kc] = ld_frag(hcat + kc * 32, 2 * HE);
#pragma unroll
            for (int nt = 0; nt < 4; ++nt) {
#pragma unroll
                for (int kc = 0; kc < 4; ++kc) {
                    v16bf b = ld_frag(sWhhT + (nbase + nt * 16) * HE + kc * 32, HE);
                    acc[nt] = wmma_bf16(ah[kc], b, acc[nt]);
                }
            }
#pragma unroll
            for (int nt = 0; nt < 4; ++nt)
#pragma unroll
                for (int r = 0; r < 8; ++r)
                    gat[(mrow + r) * G4 + nbase + nt * 16 + ncol] = acc[nt][r];
        }
        __syncthreads();

        // ---- Phase E: elementwise LSTM cell (f32) ----
        for (int i = tid; i < 16 * HE; i += NT_THREADS) {
            int m = i >> 7, h = i & 127;
            float ig = gat[m * G4 + h]            + sb[h];
            float fg = gat[m * G4 + HE + h]       + sb[HE + h];
            float gg = gat[m * G4 + 2 * HE + h]   + sb[2 * HE + h];
            float og = gat[m * G4 + 3 * HE + h]   + sb[3 * HE + h];
            float cn = fsig(fg) * cst[i] + fsig(ig) * ftanh(gg);
            float hn = fsig(og) * ftanh(cn);
            cst[i] = cn;
            hcat[m * 2 * HE + h]      = (__bf16)hn;
            hcat[m * 2 * HE + HE + h] = (__bf16)cn;
            enc_hs[((size_t)(b0 + m) * TT + t) * HE + h] = (__bf16)hn;
        }
        __syncthreads();
    }
}

// ---------------------------------------------------------------------------
// Decoder: persistent kernel, 16 batch rows per WG, t = 0..T-2.
// ---------------------------------------------------------------------------
__global__ __launch_bounds__(NT_THREADS, 1)
void dec_kernel(const float* __restrict__ X,
                const int*   __restrict__ idxp,
                const __bf16* __restrict__ gWhhT,   // [512][128]
                const float*  __restrict__ g_wih,   // [512] (1 x 4HD)
                const float*  __restrict__ g_b,     // [512]
                float*  __restrict__ dF32,          // [B][128]
                __bf16* __restrict__ dBF)           // [B][128]
{
    __shared__ __align__(16) __bf16 sWhhT[G4 * HD];
    __shared__ __align__(16) __bf16 hbuf [16 * HD];
    __shared__ __align__(16) float  hF   [16 * HD];
    __shared__ __align__(16) float  cst  [16 * HD];
    __shared__ __align__(16) float  gat  [16 * G4];
    __shared__ __align__(16) float  sb   [G4];
    __shared__ __align__(16) float  swih [G4];
    __shared__ __align__(16) float  yv   [16];

    const int tid = threadIdx.x;
    const int w   = tid >> 5;
    const int l   = tid & 31;
    const int b0  = blockIdx.x * 16;

    cp_bf16(sWhhT, gWhhT, G4 * HD);
    cp_f32(sb, g_b, G4);
    cp_f32(swih, g_wih, G4);
    for (int i = tid; i < 16 * HD; i += NT_THREADS) {
        hbuf[i] = (__bf16)0.0f; hF[i] = 0.0f; cst[i] = 0.0f;
    }
    __syncthreads();

    const int idx  = idxp[0];
    const int mrow = 8 * (l >> 4);
    const int ncol = l & 15;
    const v8f vzero = {0.f,0.f,0.f,0.f,0.f,0.f,0.f,0.f};

    for (int t = 0; t < TT - 1; ++t) {
        if (tid < 16) {
            yv[tid] = X[((size_t)(b0 + tid) * TT + t) * FF + idx];
            if (t + 2 < TT)
                __builtin_prefetch(&X[((size_t)(b0 + tid) * TT + (t + 1)) * FF + idx], 0, 0);
        }

        {   // gates = h @ Whh ; each wave owns 64 columns
            const int nbase = w * 64;
            v8f acc[4];
#pragma unroll
            for (int i = 0; i < 4; ++i) acc[i] = vzero;
            v16bf ah[4];
#pragma unroll
            for (int kc = 0; kc < 4; ++kc) ah[kc] = ld_frag(hbuf + kc * 32, HD);
#pragma unroll
            for (int nt = 0; nt < 4; ++nt) {
#pragma unroll
                for (int kc = 0; kc < 4; ++kc) {
                    v16bf b = ld_frag(sWhhT + (nbase + nt * 16) * HD + kc * 32, HD);
                    acc[nt] = wmma_bf16(ah[kc], b, acc[nt]);
                }
            }
#pragma unroll
            for (int nt = 0; nt < 4; ++nt)
#pragma unroll
                for (int r = 0; r < 8; ++r)
                    gat[(mrow + r) * G4 + nbase + nt * 16 + ncol] = acc[nt][r];
        }
        __syncthreads();

        for (int i = tid; i < 16 * HD; i += NT_THREADS) {
            int m = i >> 7, h = i & 127;
            float y  = yv[m];
            float ig = gat[m * G4 + h]          + sb[h]          + y * swih[h];
            float fg = gat[m * G4 + HD + h]     + sb[HD + h]     + y * swih[HD + h];
            float gg = gat[m * G4 + 2*HD + h]   + sb[2*HD + h]   + y * swih[2*HD + h];
            float og = gat[m * G4 + 3*HD + h]   + sb[3*HD + h]   + y * swih[3*HD + h];
            float cn = fsig(fg) * cst[i] + fsig(ig) * ftanh(gg);
            float hn = fsig(og) * ftanh(cn);
            cst[i] = cn; hF[i] = hn; hbuf[i] = (__bf16)hn;
        }
        __syncthreads();
    }

    for (int i = tid; i < 16 * HD; i += NT_THREADS) {
        int m = i >> 7, h = i & 127;
        dF32[(size_t)(b0 + m) * HD + h] = hF[i];
        dBF [(size_t)(b0 + m) * HD + h] = (__bf16)hF[i];
    }
}

// ---------------------------------------------------------------------------
// Temporal attention + LayerNorm + heads. 16 batch rows per WG.
// ---------------------------------------------------------------------------
__global__ __launch_bounds__(NT_THREADS, 1)
void head_kernel(const __bf16* __restrict__ enc_hs,   // [B][T][128] bf16
                 const float*  __restrict__ dF32g,    // [B][128]
                 const __bf16* __restrict__ dBFg,     // [B][128]
                 const __bf16* __restrict__ gW1T,     // [128][256]
                 const float*  __restrict__ g_ab1,    // [128]
                 const float*  __restrict__ g_w2,     // [128] (dec_a_w2)
                 const float*  __restrict__ g_gamma,  // [256]
                 const float*  __restrict__ g_beta,   // [256]
                 const float*  __restrict__ g_regw,   // [256]
                 const float*  __restrict__ g_regb,   // [1]
                 const float*  __restrict__ g_clsw,   // [256][3]
                 const float*  __restrict__ g_clsb,   // [3]
                 float* __restrict__ out)
{
    __shared__ __align__(16) __bf16 sW1T[HE * 2 * HE];
    __shared__ __align__(16) __bf16 dloc[16 * HD];
    __shared__ __align__(16) float  dF  [16 * HD];
    __shared__ __align__(16) float  u   [16 * HE];
    __shared__ __align__(16) float  sc  [16 * TT];
    __shared__ __align__(16) float  ctx [16 * HE];
    __shared__ __align__(16) float  w2  [HE];
    __shared__ __align__(16) float  ab1 [HE];
    __shared__ __align__(16) float  gam [2 * HE];
    __shared__ __align__(16) float  bet [2 * HE];
    __shared__ __align__(16) float  rgw [2 * HE];
    __shared__ __align__(16) float  clw [2 * HE * 3];

    const int tid = threadIdx.x;
    const int w   = tid >> 5;
    const int l   = tid & 31;
    const int b0  = blockIdx.x * 16;

    cp_bf16(sW1T, gW1T, HE * 2 * HE);
    cp_bf16(dloc, dBFg + (size_t)b0 * HD, 16 * HD);
    cp_f32(dF,  dF32g + (size_t)b0 * HD, 16 * HD);
    cp_f32(w2,  g_w2, HE);
    cp_f32(ab1, g_ab1, HE);
    cp_f32(gam, g_gamma, 2 * HE);
    cp_f32(bet, g_beta, 2 * HE);
    cp_f32(rgw, g_regw, 2 * HE);
    cp_f32(clw, g_clsw, 2 * HE * 3);
    __syncthreads();

    const int mrow = 8 * (l >> 4);
    const int ncol = l & 15;
    const v8f vzero = {0.f,0.f,0.f,0.f,0.f,0.f,0.f,0.f};

    // u = d @ W1_top  (16x128 @ 128x128), K-offset 0 of W1T
    {
        const int nt = w;   // 8 waves, 8 tiles
        v8f acc = vzero;
#pragma unroll
        for (int kc = 0; kc < 4; ++kc) {
            v16bf a = ld_frag(dloc + kc * 32, HD);
            v16bf b = ld_frag(sW1T + nt * 16 * (2 * HE) + kc * 32, 2 * HE);
            acc = wmma_bf16(a, b, acc);
        }
        int n = nt * 16 + ncol;
#pragma unroll
        for (int r = 0; r < 8; ++r)
            u[(mrow + r) * HE + n] = acc[r];
    }
    __syncthreads();

    // scores[b][t] = sum_n w2[n]*tanh( enc_h[b,t]@W1_bot + u[b] + b1 )
    for (int task = w; task < 16 * 8; task += 8) {
        const int bl = task >> 3, tt = task & 7;
        const __bf16* Abase = enc_hs + ((size_t)(b0 + bl) * TT + tt * 16) * HE;
        v16bf ah[4];
#pragma unroll
        for (int kc = 0; kc < 4; ++kc) ah[kc] = ld_frag(Abase + kc * 32, HE);
        float part[8];
#pragma unroll
        for (int r = 0; r < 8; ++r) part[r] = 0.0f;
#pragma unroll
        for (int nt = 0; nt < 8; ++nt) {
            v8f acc = vzero;
#pragma unroll
            for (int kc = 0; kc < 4; ++kc) {
                v16bf b = ld_frag(sW1T + nt * 16 * (2 * HE) + HE + kc * 32, 2 * HE);
                acc = wmma_bf16(ah[kc], b, acc);
            }
            int n = nt * 16 + ncol;
            float wn = w2[n], uu = u[bl * HE + n] + ab1[n];
#pragma unroll
            for (int r = 0; r < 8; ++r)
                part[r] += wn * ftanh(acc[r] + uu);
        }
#pragma unroll
        for (int mask = 1; mask <= 8; mask <<= 1)
#pragma unroll
            for (int r = 0; r < 8; ++r)
                part[r] += __shfl_xor(part[r], mask, 32);
        if ((l & 15) == 0) {
#pragma unroll
            for (int r = 0; r < 8; ++r)
                sc[bl * TT + tt * 16 + mrow + r] = part[r];
        }
    }
    __syncthreads();

    // softmax over T per batch row
    if (tid < 16) {
        float mx = -1e30f;
        for (int t = 0; t < TT; ++t) mx = fmaxf(mx, sc[tid * TT + t]);
        float sum = 0.0f;
        for (int t = 0; t < TT; ++t) {
            float e = __expf(sc[tid * TT + t] - mx);
            sc[tid * TT + t] = e; sum += e;
        }
        float inv = frcp(sum);
        for (int t = 0; t < TT; ++t) sc[tid * TT + t] *= inv;
    }
    __syncthreads();

    // context = sum_t beta[b,t] * enc_h[b,t,:]
    for (int i = tid; i < 16 * HE; i += NT_THREADS) {
        int bl = i >> 7, h = i & 127;
        const __bf16* eh = enc_hs + (size_t)(b0 + bl) * TT * HE + h;
        float s = 0.0f;
        for (int t = 0; t < TT; ++t)
            s += sc[bl * TT + t] * (float)eh[(size_t)t * HE];
        ctx[i] = s;
    }
    __syncthreads();

    // LayerNorm over [d, context] (256) + regression + classification heads
    if (tid < 16) {
        int m = tid, b = b0 + m;
        float mean = 0.0f;
        for (int k = 0; k < HD; ++k) mean += dF[m * HD + k];
        for (int k = 0; k < HE; ++k) mean += ctx[m * HE + k];
        mean *= (1.0f / 256.0f);
        float var = 0.0f;
        for (int k = 0; k < HD; ++k) { float d = dF[m * HD + k] - mean; var += d * d; }
        for (int k = 0; k < HE; ++k) { float d = ctx[m * HE + k] - mean; var += d * d; }
        var *= (1.0f / 256.0f);
        float inv = rsqrtf(var + 1e-5f);
        float reg = g_regb[0];
        float c0 = g_clsb[0], c1 = g_clsb[1], c2 = g_clsb[2];
        for (int k = 0; k < 2 * HE; ++k) {
            float v = (k < HD) ? dF[m * HD + k] : ctx[m * HE + (k - HD)];
            float nrm = (v - mean) * inv * gam[k] + bet[k];
            reg += nrm * rgw[k];
            c0 += nrm * clw[k * 3 + 0];
            c1 += nrm * clw[k * 3 + 1];
            c2 += nrm * clw[k * 3 + 2];
        }
        out[b] = reg;
        out[BQ + b * 3 + 0] = c0;
        out[BQ + b * 3 + 1] = c1;
        out[BQ + b * 3 + 2] = c2;
    }
}

// ---------------------------------------------------------------------------
// Launch
// ---------------------------------------------------------------------------
extern "C" void kernel_launch(void* const* d_in, const int* in_sizes, int n_in,
                              void* d_out, int out_size, void* d_ws, size_t ws_size,
                              hipStream_t stream)
{
    const float* X      = (const float*)d_in[0];
    const int*   idxp   = (const int*)  d_in[1];
    const float* encWih = (const float*)d_in[2];
    const float* encWhh = (const float*)d_in[3];
    const float* encb   = (const float*)d_in[4];
    const float* encaw1 = (const float*)d_in[5];
    const float* encab1 = (const float*)d_in[6];
    const float* encaw2 = (const float*)d_in[7];
    const float* encab2 = (const float*)d_in[8];
    const float* decWih = (const float*)d_in[9];
    const float* decWhh = (const float*)d_in[10];
    const float* decb   = (const float*)d_in[11];
    const float* decaw1 = (const float*)d_in[12];
    const float* decab1 = (const float*)d_in[13];
    const float* decaw2 = (const float*)d_in[14];
    // d_in[15] = dec_a_b2 (constant offset before softmax; no effect)
    const float* gamma  = (const float*)d_in[16];
    const float* beta   = (const float*)d_in[17];
    const float* regw   = (const float*)d_in[18];
    const float* regb   = (const float*)d_in[19];
    const float* clsw   = (const float*)d_in[20];
    const float* clsb   = (const float*)d_in[21];
    float* out = (float*)d_out;

    char* ws = (char*)d_ws;
    __bf16* enc_hs  = (__bf16*)(ws + 0);           // 67,108,864 B
    float*  dF32    = (float*) (ws + 67108864);    //  1,048,576 B
    __bf16* dBF     = (__bf16*)(ws + 68157440);    //    524,288 B
    __bf16* encWhhT = (__bf16*)(ws + 68681728);    //    131,072 B
    __bf16* encWihT = (__bf16*)(ws + 68812800);    //     32,768 B
    __bf16* encW1T  = (__bf16*)(ws + 68845568);    //     65,536 B
    __bf16* encW2T  = (__bf16*)(ws + 68911104);    //      8,192 B
    __bf16* decWhhT = (__bf16*)(ws + 68919296);    //    131,072 B
    __bf16* decW1T  = (__bf16*)(ws + 69050368);    //     65,536 B

    auto tp = [&](const float* W, __bf16* WT, int K, int N) {
        int n = K * N;
        transpose_bf16_kernel<<<(n + 255) / 256, 256, 0, stream>>>(W, WT, K, N);
    };
    tp(encWih, encWihT, FF,     G4);   // (32,512)  -> [512][32]
    tp(encWhh, encWhhT, HE,     G4);   // (128,512) -> [512][128]
    tp(encaw1, encW1T,  2 * HE, HE);   // (256,128) -> [128][256]
    tp(encaw2, encW2T,  HE,     FF);   // (128,32)  -> [32][128]
    tp(decWhh, decWhhT, HD,     G4);   // (128,512) -> [512][128]
    tp(decaw1, decW1T,  2 * HE, HE);   // (256,128) -> [128][256]

    enc_kernel<<<BQ / 16, NT_THREADS, 0, stream>>>(X, encWihT, encWhhT, encW1T, encW2T,
                                                   encb, encab1, encab2, enc_hs);
    dec_kernel<<<BQ / 16, NT_THREADS, 0, stream>>>(X, idxp, decWhhT, decWih, decb,
                                                   dF32, dBF);
    head_kernel<<<BQ / 16, NT_THREADS, 0, stream>>>(enc_hs, dF32, dBF, decW1T,
                                                    decab1, decaw2, gamma, beta,
                                                    regw, regb, clsw, clsb, out);
}
